// UniSAGEConv2_25048249270509
// MI455X (gfx1250) — matrix operand order repaired and checked by reading, not compiled, not run on hardware
//
#include <hip/hip_runtime.h>
#include <hip/hip_bf16.h>

// ---------------------------------------------------------------------------
// UniSAGEConv-style hypergraph op for MI455X (gfx1250, wave32, WMMA).
// All GEMMs run on v_wmma_f32_16x16x32_bf16 (bf16 in, f32 accumulate).
// ---------------------------------------------------------------------------

#define N_NODES   30000
#define EH        2048
#define C_IN      128
#define C         128
#define NE        32768
#define NHE       500000
#define KPAD      30016          // N_NODES padded to a multiple of 32 (K dim of H@Xp)
#define NTOT      (N_NODES + EH) // 32048

typedef __attribute__((ext_vector_type(16))) __bf16 v16bf;
typedef __attribute__((ext_vector_type(8)))  float  v8f;

union F16x16 { v16bf v; uint4 q[2]; };
union BF8    { __bf16 h[8]; uint4 q; };

// D = A(16x32 bf16) * B(32x16 bf16) + C(16x16 f32)
__device__ __forceinline__ v8f wmma_bf16(v16bf a, v16bf b, v8f c) {
    return __builtin_amdgcn_wmma_f32_16x16x32_bf16(
        /*neg_a=*/false, a, /*neg_b=*/false, b,
        /*c_mod=*/(short)0, c, /*reuse_a=*/false, /*reuse_b=*/false);
}

// 16 contiguous bf16 (two 16B chunks) -> B/A fragment half-pairs
__device__ __forceinline__ v16bf ld_bf16_2x8(const __bf16* p0, const __bf16* p1) {
    F16x16 u;
    u.q[0] = *(const uint4*)p0;
    u.q[1] = *(const uint4*)p1;
    return u.v;
}
__device__ __forceinline__ v16bf ld_bf16_16(const __bf16* p) {
    return ld_bf16_2x8(p, p + 8);
}

// 8 floats @p0 and 8 floats @p1 -> bf16 fragment; optionally accumulate sum
__device__ __forceinline__ v16bf cvt_f32x16(const float* p0, const float* p1, float* sum) {
    float4 a = *(const float4*)(p0),     b = *(const float4*)(p0 + 4);
    float4 c = *(const float4*)(p1),     d = *(const float4*)(p1 + 4);
    if (sum) {
        *sum += a.x + a.y + a.z + a.w + b.x + b.y + b.z + b.w
              + c.x + c.y + c.z + c.w + d.x + d.y + d.z + d.w;
    }
    v16bf r;
    r[0]=(__bf16)a.x;  r[1]=(__bf16)a.y;  r[2]=(__bf16)a.z;  r[3]=(__bf16)a.w;
    r[4]=(__bf16)b.x;  r[5]=(__bf16)b.y;  r[6]=(__bf16)b.z;  r[7]=(__bf16)b.w;
    r[8]=(__bf16)c.x;  r[9]=(__bf16)c.y;  r[10]=(__bf16)c.z; r[11]=(__bf16)c.w;
    r[12]=(__bf16)d.x; r[13]=(__bf16)d.y; r[14]=(__bf16)d.z; r[15]=(__bf16)d.w;
    return r;
}

// ---------------------------------------------------------------------------
// Workspace layout (bytes). Buffers needing zero-init are contiguous at the end.
// ---------------------------------------------------------------------------
#define OFF_XC     ((size_t)0)                         // f32 [NTOT][C] (Xp rows 0..N-1, Xh rows N..)
#define OFF_WT     (OFF_XC    + (size_t)NTOT * C * 4)  // bf16 [C][C_IN]     (W transposed)
#define OFF_WCT    (OFF_WT    + (size_t)C * C_IN * 2)  // bf16 [C][2C]       (Wc transposed)
#define OFF_XE     (OFF_WCT   + (size_t)C * 2 * C * 2) // f32  [EH][C]
#define OFF_XEB    (OFF_XE    + (size_t)EH * C * 4)    // bf16 [EH][C]
#define OFF_Z0     (OFF_XEB   + (size_t)EH * C * 2)    // ---- zeroed block starts ----
#define OFF_XPT    (OFF_Z0)                            // bf16 [C][KPAD]     (Xp transposed, padded)
#define OFF_XHSUM  (OFF_XPT   + (size_t)C * KPAD * 2)  // f32 [EH][C]
#define OFF_XHCNT  (OFF_XHSUM + (size_t)EH * C * 4)    // f32 [EH]
#define OFF_XVSUM  (OFF_XHCNT + (size_t)EH * 4)        // f32 [NTOT][C]
#define OFF_XVCNT  (OFF_XVSUM + (size_t)NTOT * C * 4)  // f32 [NTOT]
#define OFF_RED    (OFF_XVCNT + (size_t)NTOT * 4)      // f32 [4][C]: dot, ni2, nj2, scale
#define OFF_END    (OFF_RED   + (size_t)4 * C * 4)
#define ZERO_FLTS  ((int)((OFF_END - OFF_Z0) / 4))

// ---------------------------------------------------------------------------
__global__ void k_zero(float* __restrict__ p, int n) {
    for (int i = blockIdx.x * blockDim.x + threadIdx.x; i < n; i += gridDim.x * blockDim.x)
        p[i] = 0.0f;
}

// Transpose W [C_IN][C] -> WT bf16 [C][C_IN]; Wc [2C][C] -> WcT bf16 [C][2C]
__global__ void k_prep_w(const float* __restrict__ W, const float* __restrict__ Wc,
                         __bf16* __restrict__ WT, __bf16* __restrict__ WcT) {
    int t = blockIdx.x * blockDim.x + threadIdx.x;
    if (t < C * C_IN) {
        int n = t >> 7, k = t & 127;
        WT[n * C_IN + k] = (__bf16)W[k * C + n];
    }
    int t2 = t - C * C_IN;
    if (t2 >= 0 && t2 < C * 2 * C) {
        int n = t2 >> 8, k = t2 & 255;
        WcT[n * (2 * C) + k] = (__bf16)Wc[k * C + n];
    }
}

// ---------------------------------------------------------------------------
// GEMM 1: Xp = X @ W.  One wave per 16x128 output stripe.
// Writes Xp (f32, into Xc rows 0..N-1) and XpT (bf16, transposed, for GEMM 2).
// ---------------------------------------------------------------------------
__global__ __launch_bounds__(256) void k_gemm_xp(
        const float* __restrict__ X, const __bf16* __restrict__ WT,
        float* __restrict__ Xc, __bf16* __restrict__ XpT) {
    int wave = threadIdx.x >> 5, lane = threadIdx.x & 31;
    int tile = blockIdx.x * 8 + wave;              // 16-row tile, 1875 total
    if (tile >= N_NODES / 16) return;
    int m = lane & 15, h = lane >> 4;

    const float* xrow = X + (size_t)(tile * 16 + m) * C_IN;
    v8f acc[8] = {};
    for (int k0 = 0; k0 < C_IN; k0 += 32) {
        v16bf a = cvt_f32x16(xrow + k0 + 16 * h, xrow + k0 + 16 * h + 8, nullptr);
#pragma unroll
        for (int nt = 0; nt < 8; nt++) {
            v16bf b = ld_bf16_16(WT + (size_t)(nt * 16 + m) * C_IN + k0 + 16 * h);
            acc[nt] = wmma_bf16(a, b, acc[nt]);
        }
    }
    int row0 = tile * 16 + 8 * h;                  // C-fragment rows row0..row0+7
#pragma unroll
    for (int nt = 0; nt < 8; nt++) {
        int n = nt * 16 + m;
        BF8 pk;
#pragma unroll
        for (int r = 0; r < 8; r++) {
            float v = acc[nt][r];
            Xc[(size_t)(row0 + r) * C + n] = v;
            pk.h[r] = (__bf16)v;
        }
        *(uint4*)(XpT + (size_t)n * KPAD + row0) = pk.q;  // contiguous bf16x8
    }
}

// ---------------------------------------------------------------------------
// GEMM 2: Xe = (H @ Xp) / (rowsum(H)+0.01).  Block = 8 waves; B panel (32x128
// bf16 from XpT) staged through LDS and shared by all 8 waves; row-sum of H
// fused into the A-fragment loads (H streamed from HBM exactly once).
// ---------------------------------------------------------------------------
__global__ __launch_bounds__(256) void k_gemm_xe(
        const float* __restrict__ H, const __bf16* __restrict__ XpT,
        float* __restrict__ Xe, __bf16* __restrict__ XeB) {
    __shared__ __bf16 PB[128 * 32];     // [n][k] panel, 8 KB
    __shared__ float  RS[8][16];        // per-wave row sums

    int tid = threadIdx.x, wave = tid >> 5, lane = tid & 31;
    int m = lane & 15, h = lane >> 4;
    int tile = blockIdx.x * 8 + wave;   // 0..127, 16 blocks exactly

    const float* hrow = H + (size_t)(tile * 16 + m) * N_NODES;

    int sn = tid >> 1, sh = tid & 1;    // panel staging role: channel sn, half sh
    const __bf16* sp = XpT + (size_t)sn * KPAD + sh * 16;
    __bf16* sl = PB + sn * 32 + sh * 16;

    v8f acc[8] = {};
    float hs = 0.0f;

    for (int k0 = 0; k0 < KPAD; k0 += 32) {
        // stage B panel into LDS (zero-padded region of XpT covers the K tail)
        *(uint4*)(sl)     = *(const uint4*)(sp + k0);
        *(uint4*)(sl + 8) = *(const uint4*)(sp + k0 + 8);
        __syncthreads();

        v16bf a;
        if (k0 + 32 <= N_NODES) {
            a = cvt_f32x16(hrow + k0 + 16 * h, hrow + k0 + 16 * h + 8, &hs);
            if (k0 + 64 <= N_NODES)
                __builtin_prefetch(hrow + k0 + 32 + 8 * h, 0, 1);  // global_prefetch_b8
        } else {
            for (int i = 0; i < 16; i++) {
                int k = k0 + 16 * h + i;
                float v = (k < N_NODES) ? hrow[k] : 0.0f;
                hs += v;
                a[i] = (__bf16)v;
            }
        }
#pragma unroll
        for (int nt = 0; nt < 8; nt++) {
            v16bf b = ld_bf16_16(PB + (size_t)(nt * 16 + m) * 32 + 16 * h);
            acc[nt] = wmma_bf16(a, b, acc[nt]);
        }
        __syncthreads();
    }

    // row m total = halves h=0 and h=1 combined (cross-half-lane shuffle)
    hs += __shfl_xor(hs, 16, 32);
    if (h == 0) RS[wave][m] = hs;
    __syncthreads();

#pragma unroll
    for (int nt = 0; nt < 8; nt++) {
        int n = nt * 16 + m;
#pragma unroll
        for (int r = 0; r < 8; r++) {
            int e = tile * 16 + 8 * h + r;
            float v = acc[nt][r] / (RS[wave][8 * h + r] + 0.01f);
            Xe[(size_t)e * C + n]  = v;
            XeB[(size_t)e * C + n] = (__bf16)v;
        }
    }
}

// Xh scatter-mean denominators: count edges per dst hyperedge
__global__ void k_cnt_xh(const int* __restrict__ E, float* __restrict__ cnt) {
    int e = blockIdx.x * blockDim.x + threadIdx.x;
    if (e < NE) atomicAdd(&cnt[E[NE + e]], 1.0f);
}

// ---------------------------------------------------------------------------
// GEMM 3 (ContrastConv): msg = Xe[dst]@Wc_top + Xe[src]@Wc_bot + Xe[dst],
// atomically scattered into XhSum by dst. One wave per 16-edge tile.
// ---------------------------------------------------------------------------
__global__ __launch_bounds__(256) void k_contrast(
        const __bf16* __restrict__ XeB, const float* __restrict__ Xe,
        const __bf16* __restrict__ WcT, const int* __restrict__ E,
        float* __restrict__ XhSum) {
    int wave = threadIdx.x >> 5, lane = threadIdx.x & 31;
    int m = lane & 15, h = lane >> 4;
    int tile = blockIdx.x * 8 + wave;   // 0..2047, 256 blocks exactly
    const int* dstp = E + NE;

    int et = tile * 16 + m;             // A-fragment row owned by this lane
    int di = dstp[et], si = E[et];
    const __bf16* xi = XeB + (size_t)di * C;
    const __bf16* xj = XeB + (size_t)si * C;

    v8f acc[8] = {};
    for (int k0 = 0; k0 < C; k0 += 32) {
        v16bf a1 = ld_bf16_2x8(xi + k0 + 16 * h, xi + k0 + 16 * h + 8);
        v16bf a2 = ld_bf16_2x8(xj + k0 + 16 * h, xj + k0 + 16 * h + 8);
#pragma unroll
        for (int nt = 0; nt < 8; nt++) {
            const __bf16* wr = WcT + (size_t)(nt * 16 + m) * (2 * C);
            v16bf b1 = ld_bf16_16(wr + k0 + 16 * h);
            v16bf b2 = ld_bf16_16(wr + C + k0 + 16 * h);
            acc[nt] = wmma_bf16(a1, b1, acc[nt]);
            acc[nt] = wmma_bf16(a2, b2, acc[nt]);
        }
    }
    int rowDst[8];
#pragma unroll
    for (int r = 0; r < 8; r++) rowDst[r] = dstp[tile * 16 + 8 * h + r];
#pragma unroll
    for (int nt = 0; nt < 8; nt++) {
        int n = nt * 16 + m;
#pragma unroll
        for (int r = 0; r < 8; r++) {
            int d = rowDst[r];
            float v = acc[nt][r] + Xe[(size_t)d * C + n];   // "+ x_i" term in f32
            atomicAdd(&XhSum[(size_t)d * C + n], v);
        }
    }
}

// Xh = XhSum / max(cnt,1), written into Xc rows N..N+EH-1
__global__ void k_fin_xh(const float* __restrict__ XhSum, const float* __restrict__ cnt,
                         float* __restrict__ Xc) {
    int t = blockIdx.x * blockDim.x + threadIdx.x;
    if (t < EH * C) {
        int e = t >> 7;
        Xc[(size_t)N_NODES * C + t] = XhSum[t] / fmaxf(cnt[e], 1.0f);
    }
}

// ---------------------------------------------------------------------------
// Att_Agg pass 1: per-channel dot / ||xi||^2 / ||xj||^2 over all NHE edges.
// 128 threads = one channel each; block-local accumulation then 3 atomics.
// ---------------------------------------------------------------------------
__global__ __launch_bounds__(128) void k_att_dot(
        const float* __restrict__ Xc, const int* __restrict__ ehe,
        float* __restrict__ red) {
    int c = threadIdx.x;
    const int* s2 = ehe;
    const int* d2 = ehe + NHE;
    float sd = 0.f, si = 0.f, sj = 0.f;
    for (int e = blockIdx.x; e < NHE; e += gridDim.x) {
        int di = d2[e], so = s2[e];
        float xi = Xc[(size_t)di * C + c];
        float xj = Xc[(size_t)so * C + c];
        sd += xi * xj; si += xi * xi; sj += xj * xj;
    }
    atomicAdd(&red[c], sd);
    atomicAdd(&red[C + c], si);
    atomicAdd(&red[2 * C + c], sj);
}

__global__ void k_att_cos(float* __restrict__ red) {
    int c = threadIdx.x;
    float d = red[c], a = red[C + c], b = red[2 * C + c];
    red[3 * C + c] = 1.0f - d / fmaxf(sqrtf(a * b), 1e-8f);   // (1 - cos)
}

// Att_Agg pass 2: t = (1-cos)*xj scattered by d2 (sum + count for mean)
__global__ __launch_bounds__(128) void k_att_scatter(
        const float* __restrict__ Xc, const int* __restrict__ ehe,
        const float* __restrict__ red, float* __restrict__ XvSum,
        float* __restrict__ XvCnt) {
    int c = threadIdx.x;
    const int* s2 = ehe;
    const int* d2 = ehe + NHE;
    float sc = red[3 * C + c];
    for (int e = blockIdx.x; e < NHE; e += gridDim.x) {
        int di = d2[e], so = s2[e];
        float xj = Xc[(size_t)so * C + c];
        atomicAdd(&XvSum[(size_t)di * C + c], sc * xj);
        if (c == 0) atomicAdd(&XvCnt[di], 1.0f);
    }
}

// out = Xp + Xv[index]
__global__ void k_out(const float* __restrict__ Xc, const float* __restrict__ XvSum,
                      const float* __restrict__ XvCnt, const int* __restrict__ index,
                      float* __restrict__ out) {
    int t = blockIdx.x * blockDim.x + threadIdx.x;
    if (t < N_NODES * C) {
        int i = t >> 7, c = t & 127;
        int g = index[i];
        out[t] = Xc[t] + XvSum[(size_t)g * C + c] / fmaxf(XvCnt[g], 1.0f);
    }
}

// ---------------------------------------------------------------------------
extern "C" void kernel_launch(void* const* d_in, const int* in_sizes, int n_in,
                              void* d_out, int out_size, void* d_ws, size_t ws_size,
                              hipStream_t stream) {
    const float* X   = (const float*)d_in[0];
    const float* H   = (const float*)d_in[1];
    const int*   E   = (const int*)d_in[2];
    const int*   EHE = (const int*)d_in[3];
    const int*   IDX = (const int*)d_in[4];
    const float* W   = (const float*)d_in[5];
    const float* Wc  = (const float*)d_in[6];
    float* out = (float*)d_out;

    char* ws = (char*)d_ws;
    if (ws_size < OFF_END) return;   // workspace layout requires ~41.4 MiB

    float*  Xc    = (float*) (ws + OFF_XC);
    __bf16* WT    = (__bf16*)(ws + OFF_WT);
    __bf16* WcT   = (__bf16*)(ws + OFF_WCT);
    float*  Xe    = (float*) (ws + OFF_XE);
    __bf16* XeB   = (__bf16*)(ws + OFF_XEB);
    __bf16* XpT   = (__bf16*)(ws + OFF_XPT);
    float*  XhSum = (float*) (ws + OFF_XHSUM);
    float*  XhCnt = (float*) (ws + OFF_XHCNT);
    float*  XvSum = (float*) (ws + OFF_XVSUM);
    float*  XvCnt = (float*) (ws + OFF_XVCNT);
    float*  red   = (float*) (ws + OFF_RED);

    k_zero<<<4096, 256, 0, stream>>>((float*)(ws + OFF_Z0), ZERO_FLTS);
    k_prep_w<<<(C * C_IN + C * 2 * C + 255) / 256, 256, 0, stream>>>(W, Wc, WT, WcT);
    k_gemm_xp<<<(N_NODES / 16 + 7) / 8, 256, 0, stream>>>(X, WT, Xc, XpT);
    k_gemm_xe<<<EH / 16 / 8, 256, 0, stream>>>(H, XpT, Xe, XeB);
    k_cnt_xh<<<(NE + 255) / 256, 256, 0, stream>>>(E, XhCnt);
    k_contrast<<<NE / 16 / 8, 256, 0, stream>>>(XeB, Xe, WcT, E, XhSum);
    k_fin_xh<<<(EH * C + 255) / 256, 256, 0, stream>>>(XhSum, XhCnt, Xc);
    k_att_dot<<<1024, 128, 0, stream>>>(Xc, EHE, red);
    k_att_cos<<<1, 128, 0, stream>>>(red);
    k_att_scatter<<<2048, 128, 0, stream>>>(Xc, EHE, red, XvSum, XvCnt);
    k_out<<<(N_NODES * C + 255) / 256, 256, 0, stream>>>(Xc, XvSum, XvCnt, IDX, out);
}
